// NetTGCN_67070209295119
// MI455X (gfx1250) — compile-verified, matrix operand order checked by reading, not run on hardware
//
#include <hip/hip_runtime.h>
#include <hip/hip_bf16.h>
#include <math.h>

typedef __attribute__((ext_vector_type(16))) _Float16 v16h;
typedef __attribute__((ext_vector_type(8)))  float    v8f;

#define B_   32
#define N0_  4096
#define T_   30
#define K_   25
#define G1_  32
#define G2_  64
#define D_   512
#define C_   10
#define E0_  131072
#define E2_  32768
#define N2_  1024

// ---------------------------------------------------------------- utilities

__global__ void k_zero(float* __restrict__ p, size_t n) {
    size_t i = (size_t)blockIdx.x * blockDim.x + threadIdx.x;
    if (i < n) p[i] = 0.0f;
}

__global__ void k_deg(const int* __restrict__ row, int E, float* __restrict__ deg) {
    int i = blockIdx.x * blockDim.x + threadIdx.x;
    if (i < E) atomicAdd(&deg[row[i]], 1.0f);
}

__global__ void k_edgew(const int* __restrict__ row, const int* __restrict__ col,
                        const float* __restrict__ deg, float* __restrict__ w, int E) {
    int i = blockIdx.x * blockDim.x + threadIdx.x;
    if (i >= E) return;
    float dr = deg[row[i]], dc = deg[col[i]];
    float ir = (dr > 0.0f) ? rsqrtf(fmaxf(dr, 1.0f)) : 0.0f;
    float ic = (dc > 0.0f) ? rsqrtf(fmaxf(dc, 1.0f)) : 0.0f;
    w[i] = -ir * ic;
}

// exclusive prefix scan of (int)deg[0..n) -> rowptr[0..n]; single 1024-thread block,
// up to 4 elements per thread, LDS Hillis-Steele across thread partials.
__global__ void k_scan(const float* __restrict__ deg, int* __restrict__ rowptr, int n) {
    __shared__ int s[1024];
    int t = threadIdx.x;
    int base = t * 4;
    int local[4];
    int sum = 0;
#pragma unroll
    for (int j = 0; j < 4; ++j) {
        int v = (base + j < n) ? (int)deg[base + j] : 0;
        local[j] = sum;
        sum += v;
    }
    s[t] = sum;
    __syncthreads();
    for (int off = 1; off < 1024; off <<= 1) {
        int v = (t >= off) ? s[t - off] : 0;
        __syncthreads();
        s[t] += v;
        __syncthreads();
    }
    int pre = (t > 0) ? s[t - 1] : 0;
#pragma unroll
    for (int j = 0; j < 4; ++j)
        if (base + j < n) rowptr[base + j] = pre + local[j];
    if (t == 1023) rowptr[n] = s[1023];
}

// scatter edge ids into CSR slots
__global__ void k_scatter(const int* __restrict__ row, int E, const int* __restrict__ rowptr,
                          int* __restrict__ fill, int* __restrict__ eidx) {
    int i = blockIdx.x * blockDim.x + threadIdx.x;
    if (i >= E) return;
    int r = row[i];
    int p = atomicAdd(&fill[r], 1);
    eidx[rowptr[r] + p] = i;
}

// real part of FFT over time axis: out[(n*B+b)*T + f] = sum_t x[b,n,t]*cos(2*pi*f*t/T)
__global__ void k_fftreal(const float* __restrict__ x, float* __restrict__ o) {
    size_t i = (size_t)blockIdx.x * blockDim.x + threadIdx.x;
    if (i >= (size_t)N0_ * B_ * T_) return;
    int    f = (int)(i % T_);
    int    b = (int)((i / T_) % B_);
    size_t n = i / ((size_t)T_ * B_);
    const float* xs = x + ((size_t)b * N0_ + n) * T_;
    float w = 6.283185307179586f * (float)f / (float)T_;
    float s = 0.0f;
    for (int t = 0; t < T_; ++t) s += xs[t] * cosf(w * (float)t);
    o[i] = s;
}

// CSR gather SpMV fused with Chebyshev combine:
//   dst[n,f] = alpha * sum_{e in row n} w[e]*src[col[e],f] + beta*prev[n,f]
// One block per destination row; (col,w) staged in LDS; no atomics.
__global__ void k_spmv_csr(const int* __restrict__ rowptr, const int* __restrict__ eidx,
                           const int* __restrict__ col, const float* __restrict__ w,
                           const float* __restrict__ src, const float* __restrict__ prev,
                           float* __restrict__ dst, float alpha, float beta, int F) {
    __shared__ int   s_c[256];
    __shared__ float s_w[256];
    int n = blockIdx.x;
    int f = blockIdx.y * blockDim.x + threadIdx.x;
    int start = rowptr[n], end = rowptr[n + 1];
    float acc = 0.0f;
    for (int base = start; base < end; base += 256) {
        int m = end - base;
        if (m > 256) m = 256;
        if (threadIdx.x < m) {
            int e = eidx[base + threadIdx.x];
            s_c[threadIdx.x] = col[e];
            s_w[threadIdx.x] = w[e];
        }
        __syncthreads();
        if (f < F)
            for (int j = 0; j < m; ++j)
                acc += s_w[j] * src[(size_t)s_c[j] * F + f];
        __syncthreads();
    }
    if (f < F) {
        float r = alpha * acc;
        if (beta != 0.0f) r += beta * prev[(size_t)n * F + f];
        dst[(size_t)n * F + f] = r;
    }
}

// ------------------------------------------------- WMMA GEMM accumulate pass
// C[M,NN] += A[M,KD] * Bm[KD,NN]  (KD <= 32, zero-padded into one 16x16x32 step)
// One wave per 16-row m-tile; A fragment loaded once, reused across all NT n-tiles.
template<int KD, int NN, int NT>
__global__ void k_contract_t(const float* __restrict__ A, const float* __restrict__ Bm,
                             float* __restrict__ Cm, int M) {
    int wave = (int)((blockIdx.x * (size_t)blockDim.x + threadIdx.x) >> 5);
    int lane = threadIdx.x & 31;
    if (wave >= (M >> 4)) return;
    int half = lane >> 4;
    int mn   = lane & 15;

    const float* Arow = A + (size_t)(wave * 16 + mn) * KD;

    v16h a;
#pragma unroll
    for (int v = 0; v < 8; ++v) {
        int k0 = ((v & 3) << 1) + ((v >> 2) << 4) + (half << 3);
        float a0 = (k0     < KD) ? Arow[k0]     : 0.0f;
        float a1 = (k0 + 1 < KD) ? Arow[k0 + 1] : 0.0f;
        a[2 * v] = (_Float16)a0;  a[2 * v + 1] = (_Float16)a1;
    }

#pragma unroll
    for (int nt = 0; nt < NT; ++nt) {
        int ncol = nt * 16 + mn;
        v16h b;
#pragma unroll
        for (int v = 0; v < 8; ++v) {
            int k0 = ((v & 3) << 1) + ((v >> 2) << 4) + (half << 3);
            float b0 = (k0     < KD) ? Bm[k0 * NN + ncol]       : 0.0f;
            float b1 = (k0 + 1 < KD) ? Bm[(k0 + 1) * NN + ncol] : 0.0f;
            b[2 * v] = (_Float16)b0;  b[2 * v + 1] = (_Float16)b1;
        }
        float* Cb = Cm + (size_t)(wave * 16 + half * 8) * NN + ncol;
        v8f c;
#pragma unroll
        for (int v = 0; v < 8; ++v) c[v] = Cb[(size_t)v * NN];
        c = __builtin_amdgcn_wmma_f32_16x16x32_f16(false, a, false, b, (short)0, c, false, false);
#pragma unroll
        for (int v = 0; v < 8; ++v) Cb[(size_t)v * NN] = c[v];
    }
}

// bias + relu + max-pool over groups of 4 nodes: h1acc[N0,B,G1] -> ty0[N2,B,G1]
__global__ void k_pool(const float* __restrict__ h, const float* __restrict__ bias,
                       float* __restrict__ o) {
    size_t i = (size_t)blockIdx.x * blockDim.x + threadIdx.x;
    if (i >= (size_t)N2_ * B_ * G1_) return;
    int    g  = (int)(i % G1_);
    int    b  = (int)((i / G1_) % B_);
    size_t n2 = i / ((size_t)G1_ * B_);
    float m = -1e30f;
#pragma unroll
    for (int j = 0; j < 4; ++j)
        m = fmaxf(m, h[((4 * n2 + j) * B_ + b) * G1_ + g]);
    o[i] = fmaxf(m + bias[g], 0.0f);   // relu(max(h)+b) == max(relu(h+b))
}

// bias + relu + transpose to [B, N2*G2] for the classifier
__global__ void k_flat(const float* __restrict__ h2, const float* __restrict__ bias,
                       float* __restrict__ flat) {
    size_t i = (size_t)blockIdx.x * blockDim.x + threadIdx.x;
    if (i >= (size_t)N2_ * B_ * G2_) return;
    int    g  = (int)(i % G2_);
    int    b  = (int)((i / G2_) % B_);
    size_t n2 = i / ((size_t)G2_ * B_);
    flat[(size_t)b * (N2_ * G2_) + n2 * G2_ + g] = fmaxf(h2[i] + bias[g], 0.0f);
}

// fc1: zacc[32,512] += flat[32,65536] @ fc1_w[65536,512], K split across grid.z.
// A fragment per lane is two contiguous 8-float runs -> four b128 loads.
__global__ void k_fc1(const float* __restrict__ Aflat, const float* __restrict__ W,
                      float* __restrict__ zacc, int Ktot, int klen) {
    const float* Af = (const float*)__builtin_assume_aligned(Aflat, 16);
    int lane   = threadIdx.x & 31;
    int tile_m = blockIdx.x;               // 0..1
    int tile_n = blockIdx.y;               // 0..31
    int kbase  = blockIdx.z * klen;
    int half = lane >> 4, mn = lane & 15;
    const float* Arow = Af + (size_t)(tile_m * 16 + mn) * Ktot + kbase;
    int ncol = tile_n * 16 + mn;

    v8f c = {0.f, 0.f, 0.f, 0.f, 0.f, 0.f, 0.f, 0.f};
    for (int kk = 0; kk < klen; kk += 32) {
        if (kk + 32 < klen)
            __builtin_prefetch(&W[(size_t)(kbase + kk + 32) * D_ + ncol], 0, 0);
        const float4* A4a = (const float4*)(Arow + kk + half * 8);
        const float4* A4b = (const float4*)(Arow + kk + 16 + half * 8);
        float4 x0 = A4a[0], x1 = A4a[1];
        float4 x2 = A4b[0], x3 = A4b[1];
        v16h a;
        a[0]  = (_Float16)x0.x;  a[1]  = (_Float16)x0.y;
        a[2]  = (_Float16)x0.z;  a[3]  = (_Float16)x0.w;
        a[4]  = (_Float16)x1.x;  a[5]  = (_Float16)x1.y;
        a[6]  = (_Float16)x1.z;  a[7]  = (_Float16)x1.w;
        a[8]  = (_Float16)x2.x;  a[9]  = (_Float16)x2.y;
        a[10] = (_Float16)x2.z;  a[11] = (_Float16)x2.w;
        a[12] = (_Float16)x3.x;  a[13] = (_Float16)x3.y;
        a[14] = (_Float16)x3.z;  a[15] = (_Float16)x3.w;
        v16h b;
#pragma unroll
        for (int v = 0; v < 8; ++v) {
            int k0 = ((v & 3) << 1) + ((v >> 2) << 4) + (half << 3) + kbase + kk;
            b[2 * v]     = (_Float16)W[(size_t)k0 * D_ + ncol];
            b[2 * v + 1] = (_Float16)W[(size_t)(k0 + 1) * D_ + ncol];
        }
        c = __builtin_amdgcn_wmma_f32_16x16x32_f16(false, a, false, b, (short)0, c, false, false);
    }
    float* zb = zacc + (size_t)(tile_m * 16 + half * 8) * D_ + tile_n * 16 + mn;
#pragma unroll
    for (int v = 0; v < 8; ++v) atomicAdd(&zb[(size_t)v * D_], c[v]);
}

// relu(zacc+fc1_b) @ fc2_w + fc2_b, then log_softmax. One thread per batch row.
__global__ void k_head(const float* __restrict__ zacc, const float* __restrict__ fc1b,
                       const float* __restrict__ fc2w, const float* __restrict__ fc2b,
                       float* __restrict__ out) {
    int b = threadIdx.x;
    if (b >= B_) return;
    float l[C_];
#pragma unroll
    for (int c = 0; c < C_; ++c) l[c] = fc2b[c];
    for (int d = 0; d < D_; ++d) {
        float z = fmaxf(zacc[(size_t)b * D_ + d] + fc1b[d], 0.0f);
#pragma unroll
        for (int c = 0; c < C_; ++c) l[c] += z * fc2w[d * C_ + c];
    }
    float m = l[0];
#pragma unroll
    for (int c = 1; c < C_; ++c) m = fmaxf(m, l[c]);
    float s = 0.0f;
#pragma unroll
    for (int c = 0; c < C_; ++c) s += expf(l[c] - m);
    float ls = logf(s);
#pragma unroll
    for (int c = 0; c < C_; ++c) out[b * C_ + c] = l[c] - m - ls;
}

// ---------------------------------------------------------------- host side

static inline void zero_f(float* p, size_t n, hipStream_t s) {
    k_zero<<<dim3((unsigned)((n + 255) / 256)), 256, 0, s>>>(p, n);
}

extern "C" void kernel_launch(void* const* d_in, const int* in_sizes, int n_in,
                              void* d_out, int out_size, void* d_ws, size_t ws_size,
                              hipStream_t stream) {
    (void)in_sizes; (void)n_in; (void)out_size; (void)ws_size;

    const float* x    = (const float*)d_in[0];
    const int*   ei0  = (const int*)d_in[1];     // [2, E0]
    const int*   ei2  = (const int*)d_in[2];     // [2, E2]
    const float* W1   = (const float*)d_in[3];   // [K, T, G1]
    const float* b1   = (const float*)d_in[4];
    const float* W2   = (const float*)d_in[5];   // [K, G1, G2]
    const float* b2   = (const float*)d_in[6];
    const float* fc1w = (const float*)d_in[7];   // [N2*G2, D]
    const float* fc1b = (const float*)d_in[8];
    const float* fc2w = (const float*)d_in[9];   // [D, C]
    const float* fc2b = (const float*)d_in[10];
    float* out = (float*)d_out;

    float* w = (float*)d_ws;
    const size_t SZ1 = (size_t)N0_ * B_ * T_;    // 3,932,160
    const size_t SZ2 = (size_t)N2_ * B_ * G1_;   // 1,048,576
    // Stage-1 Chebyshev ring
    float* tx0 = w;
    float* tx1 = w + SZ1;
    float* tx2 = w + 2 * SZ1;
    // Stage-2 buffers alias the (dead by then) stage-1 ring
    float* ty0   = w;
    float* ty1   = w + SZ2;
    float* ty2   = w + 2 * SZ2;
    float* h2acc = w + 3 * SZ2;                         // [N2*B, G2]
    float* flat  = h2acc + (size_t)N2_ * B_ * G2_;      // [B, N2*G2]
    // Persistent buffers past the ring
    float* h1acc = w + 3 * SZ1;                         // [N0*B, G1]
    float* zacc  = h1acc + (size_t)N0_ * B_ * G1_;      // [B, D]
    float* deg0  = zacc + (size_t)B_ * D_;
    float* w0e   = deg0 + N0_;
    float* deg2  = w0e + E0_;
    float* w2e   = deg2 + N2_;
    int* rowptr0 = (int*)(w2e + E2_);                   // N0+1
    int* fill0   = rowptr0 + (N0_ + 1);                 // N0
    int* eidx0   = fill0 + N0_;                         // E0
    int* rowptr2 = eidx0 + E0_;                         // N2+1
    int* fill2   = rowptr2 + (N2_ + 1);                 // N2
    int* eidx2   = fill2 + N2_;                         // E2

    const int* row0 = ei0;          const int* col0 = ei0 + E0_;
    const int* row2 = ei2;          const int* col2 = ei2 + E2_;

    // ----- degrees, Laplacian edge weights, CSR build (both graphs)
    zero_f(deg0, N0_, stream);
    zero_f(deg2, N2_, stream);
    k_deg<<<(E0_ + 255) / 256, 256, 0, stream>>>(row0, E0_, deg0);
    k_deg<<<(E2_ + 255) / 256, 256, 0, stream>>>(row2, E2_, deg2);
    k_edgew<<<(E0_ + 255) / 256, 256, 0, stream>>>(row0, col0, deg0, w0e, E0_);
    k_edgew<<<(E2_ + 255) / 256, 256, 0, stream>>>(row2, col2, deg2, w2e, E2_);
    k_scan<<<1, 1024, 0, stream>>>(deg0, rowptr0, N0_);
    k_scan<<<1, 1024, 0, stream>>>(deg2, rowptr2, N2_);
    zero_f((float*)fill0, N0_, stream);
    zero_f((float*)fill2, N2_, stream);
    k_scatter<<<(E0_ + 255) / 256, 256, 0, stream>>>(row0, E0_, rowptr0, fill0, eidx0);
    k_scatter<<<(E2_ + 255) / 256, 256, 0, stream>>>(row2, E2_, rowptr2, fill2, eidx2);

    // ----- FFT real part -> tx0 [N0,B,T]
    k_fftreal<<<(unsigned)((SZ1 + 255) / 256), 256, 0, stream>>>(x, tx0);

    // ----- Stage 1: ChebTimeConv, h1acc[N0*B, G1] = sum_k T_k @ W1[k]
    const int M1 = N0_ * B_;
    const int F1 = B_ * T_;                              // 960
    const unsigned blk1 = (unsigned)(((size_t)(M1 / 16) * 32 + 255) / 256);
    const dim3 g_sp1(N0_, (F1 + 255) / 256);
    zero_f(h1acc, (size_t)N0_ * B_ * G1_, stream);

    k_contract_t<T_, G1_, 2><<<blk1, 256, 0, stream>>>(tx0, W1, h1acc, M1);

    k_spmv_csr<<<g_sp1, 256, 0, stream>>>(rowptr0, eidx0, col0, w0e,
                                          tx0, tx0, tx1, 1.0f, 0.0f, F1);
    k_contract_t<T_, G1_, 2><<<blk1, 256, 0, stream>>>(tx1, W1 + (size_t)T_ * G1_, h1acc, M1);

    float *p0 = tx0, *p1 = tx1, *p2 = tx2;
    for (int k = 2; k < K_; ++k) {
        k_spmv_csr<<<g_sp1, 256, 0, stream>>>(rowptr0, eidx0, col0, w0e,
                                              p1, p0, p2, 2.0f, -1.0f, F1);
        k_contract_t<T_, G1_, 2><<<blk1, 256, 0, stream>>>(p2, W1 + (size_t)k * T_ * G1_, h1acc, M1);
        float* t = p0; p0 = p1; p1 = p2; p2 = t;
    }

    // ----- bias + relu + pool4 -> ty0 [N2,B,G1]
    k_pool<<<(unsigned)((SZ2 + 255) / 256), 256, 0, stream>>>(h1acc, b1, ty0);

    // ----- Stage 2: ChebConv, h2acc[N2*B, G2] = sum_k T_k @ W2[k]
    const int M2 = N2_ * B_;
    const int F2 = B_ * G1_;                             // 1024
    const unsigned blk2 = (unsigned)(((size_t)(M2 / 16) * 32 + 255) / 256);
    const dim3 g_sp2(N2_, (F2 + 255) / 256);
    zero_f(h2acc, (size_t)N2_ * B_ * G2_, stream);

    k_contract_t<G1_, G2_, 4><<<blk2, 256, 0, stream>>>(ty0, W2, h2acc, M2);

    k_spmv_csr<<<g_sp2, 256, 0, stream>>>(rowptr2, eidx2, col2, w2e,
                                          ty0, ty0, ty1, 1.0f, 0.0f, F2);
    k_contract_t<G1_, G2_, 4><<<blk2, 256, 0, stream>>>(ty1, W2 + (size_t)G1_ * G2_, h2acc, M2);

    float *q0 = ty0, *q1 = ty1, *q2 = ty2;
    for (int k = 2; k < K_; ++k) {
        k_spmv_csr<<<g_sp2, 256, 0, stream>>>(rowptr2, eidx2, col2, w2e,
                                              q1, q0, q2, 2.0f, -1.0f, F2);
        k_contract_t<G1_, G2_, 4><<<blk2, 256, 0, stream>>>(q2, W2 + (size_t)k * G1_ * G2_, h2acc, M2);
        float* t = q0; q0 = q1; q1 = q2; q2 = t;
    }

    // ----- bias + relu + transpose -> flat [B, N2*G2]
    k_flat<<<(unsigned)(((size_t)N2_ * B_ * G2_ + 255) / 256), 256, 0, stream>>>(h2acc, b2, flat);

    // ----- fc1 (K-split WMMA with f32 atomic reduction), then fc2 + log_softmax
    zero_f(zacc, (size_t)B_ * D_, stream);
    dim3 gfc1(2, 32, 64);
    k_fc1<<<gfc1, 32, 0, stream>>>(flat, fc1w, zacc, N2_ * G2_, (N2_ * G2_) / 64);
    k_head<<<1, 32, 0, stream>>>(zacc, fc1b, fc2w, fc2b, out);
}
